// MinHead_25099788878663
// MI455X (gfx1250) — matrix operand order repaired and checked by reading, not compiled, not run on hardware
//
#include <hip/hip_runtime.h>
#include <hip/hip_bf16.h>
#include <math.h>

// ---------------------------------------------------------------------------
// Problem constants (from the reference)
// ---------------------------------------------------------------------------
#define SIZEC      2048
#define POOLC      10
#define KSZC       9
#define BC         8
#define TRACK_DIMC 8
#define SEQ_DIMC   4
#define DEPTHC     2
#define NCH        (TRACK_DIMC + SEQ_DIMC)   // 12 conv input channels
#define KTOT       (NCH * KSZC)              // 108 im2col K
#define PPOS       (SIZEC * POOLC)           // 20480 pre-pool positions
#define NDIAG      1022                      // len(arange(1, 1025-2))
#define TOTAL_INL  1572352L                  // sum_{d=1..1024} (2048-d)
#define OUT_PER_BL 1568259L                  // sum_{i=0..1021} (2045-i)

typedef __attribute__((ext_vector_type(16))) _Float16 v16h;
typedef __attribute__((ext_vector_type(8)))  float    v8f;

#if defined(__gfx1250__) &&                                            \
    __has_builtin(__builtin_amdgcn_global_load_async_to_lds_b32) &&    \
    __has_builtin(__builtin_amdgcn_s_wait_asynccnt)
#define HAS_ASYNC_LDS 1
typedef __attribute__((address_space(1))) int* gptr_t;   // global src (int*)
typedef __attribute__((address_space(3))) int* lptr_t;   // LDS dst (int*)
#else
#define HAS_ASYNC_LDS 0
#endif

// ---------------------------------------------------------------------------
// Kernel 1: converter conv as WMMA GEMM.
//   y[o,p] = sum_{i<12,k<9} x[ch i][p+k-4] * W[o,i,k], x = concat(seq, tracks)
//   GEMM: A(16x128 f16, rows 0..1 = filters, K = i*9+k padded 108->128)
//         B(128x16 f16, im2col tile of 16 positions, packed from an LDS-staged
//           window of 12ch x 136 positions shared by the block's 8 tiles)
//   4 chunks of V_WMMA_F32_16X16X32_F16; sigmoid(y + bias) -> workspace.
// 256-thread block = 8 wave32 tiles = 128 consecutive positions, one batch.
// ---------------------------------------------------------------------------
__global__ __launch_bounds__(256) void conv_wmma_kernel(
    const float* __restrict__ tracks,   // (8, 8, 20480)
    const float* __restrict__ seq,      // (8, 4, 20480)
    const float* __restrict__ W,        // (2, 12, 9) flat
    const float* __restrict__ bias,     // (2,)
    float* __restrict__ ypost)          // (8, 2, 20480)
{
    __shared__ float xs[NCH][136];      // 128 positions + 4 halo each side

    const int tid   = threadIdx.x;
    const int lane  = tid & 31;
    const int wtile = tid >> 5;              // 0..7: tile within block
    const int tile0 = blockIdx.x * 8;        // first global tile of block
    const int bb    = tile0 / (PPOS / 16);   // batch (no straddle: 1280 % 8 == 0)
    const int p0    = (tile0 % (PPOS / 16)) * 16;  // first position of block
    const int n     = lane & 15;             // matrix row/col within half-wave
    const int hi    = lane >> 4;             // 0: lanes 0-15, 1: lanes 16-31

    // ---- stage the shared input window (zero-padded at batch edges) -------
    for (int idx = tid; idx < NCH * 136; idx += 256) {
        int ch = idx / 136;
        int j  = idx - ch * 136;
        int sp = p0 + j - (KSZC / 2);        // SAME pad = 4
        float v = 0.0f;
        if (sp >= 0 && sp < PPOS) {
            v = (ch < SEQ_DIMC)
                    ? seq[((long)bb * SEQ_DIMC + ch) * PPOS + sp]
                    : tracks[((long)bb * TRACK_DIMC + (ch - SEQ_DIMC)) * PPOS + sp];
        }
        xs[ch][j] = v;
    }
    __syncthreads();

    v8f c = {};
#pragma unroll
    for (int chunk = 0; chunk < 4; ++chunk) {
        v16h a, b;
#pragma unroll
        for (int h = 0; h < 16; ++h) {
            // ---- A (16x32 f16 layout): K = (h<8 ? h : h+8) + 8*hi + 32*chunk
            int Ka = ((h < 8) ? h : (h + 8)) + (hi ? 8 : 0) + 32 * chunk;
            float av = 0.0f;
            if (n < 2 && Ka < KTOT) av = W[n * KTOT + Ka];
            a[h] = (_Float16)av;

            // ---- B (32x16 f16 layout): K = (h<8 ? h : h+8) + 8*hi + 32*chunk, N = n
            int Kb = ((h < 8) ? h : (h + 8)) + (hi ? 8 : 0) + 32 * chunk;
            float bv = 0.0f;
            if (Kb < KTOT) {
                int ch = Kb / KSZC;
                int kk = Kb - ch * KSZC;
                bv = xs[ch][wtile * 16 + n + kk];   // max index 135
            }
            b[h] = (_Float16)bv;
        }
        // D = A*B + C   (v_wmma_f32_16x16x32_f16)
        c = __builtin_amdgcn_wmma_f32_16x16x32_f16(
                false, a, false, b, (short)0, c, false, false);
    }

    // Rows M=0 / M=1 live in c[0] / c[1] of lanes 0..15 (N = lane).
    if (lane < 16) {
        int p = p0 + wtile * 16 + n;
        float s0 = 1.0f / (1.0f + expf(-(c[0] + bias[0])));
        float s1 = 1.0f / (1.0f + expf(-(c[1] + bias[1])));
        ypost[((long)bb * 2 + 0) * PPOS + p] = s0;
        ypost[((long)bb * 2 + 1) * PPOS + p] = s1;
    }
}

// ---------------------------------------------------------------------------
// Kernel 2: min-pool by POOL=10 along last dim: (8,2,20480) -> (8,2,2048)
// ---------------------------------------------------------------------------
__global__ __launch_bounds__(256) void pool_kernel(
    const float* __restrict__ ypost, float* __restrict__ lr)
{
    int idx = blockIdx.x * blockDim.x + threadIdx.x;
    if (idx >= BC * 2 * SIZEC) return;
    const float* src = ypost + (long)idx * POOLC;
    float m = src[0];
#pragma unroll
    for (int j = 1; j < POOLC; ++j) m = fminf(m, src[j]);
    lr[idx] = m;
}

// ---------------------------------------------------------------------------
// Kernel 3: DLEM recurrence. One block per diagonal i (1022 blocks).
// cur strip (8 x <=2047) lives entirely in LDS (64KB of the 320KB/WGP).
// Initial diagonals->LDS copy uses GLOBAL_LOAD_ASYNC_TO_LDS_B32 (ASYNCcnt),
// then exp() in place.  Each depth step: register-stage cur[k..k+16]
// (neighbor-safe in-place update), lv=log(new), block reduction of the masked
// sum (wave32 shfl_xor + LDS), normalize cur = exp(lv - m), emit lv - m on
// the final step.  left/right stream from the 192MB L2 (128KB resident).
// ---------------------------------------------------------------------------
__global__ __launch_bounds__(1024) void dlem_kernel(
    const float* __restrict__ diag,   // (8, TOTAL_IN)
    const float* __restrict__ lr,     // (8, 2, 2048): [b][0]=left, [b][1]=right
    const float* __restrict__ cptr,   // (1,) const = 0.99
    float* __restrict__ out)          // (8, OUT_PER_B)
{
    __shared__ float cur[BC][SIZEC];
    __shared__ float wsum[32];
    __shared__ float mshare;

    const int   i     = blockIdx.x;          // diagonal index, d = i+1
    const int   tid   = threadIdx.x;
    const int   b     = tid >> 7;
    const int   kbase = (tid & 127) << 4;
    const int   lens  = SIZEC - 1 - i;       // 2047 - i
    const long  off   = 2048L * i - (long)i * (i + 1) / 2;
    const float cst   = cptr[0];

    const float* leftp  = lr + ((long)b * 2 + 0) * SIZEC;
    const float* rightp = lr + ((long)b * 2 + 1) * SIZEC;
    const float* dg     = diag + (long)b * TOTAL_INL + off;

    __builtin_prefetch(dg + kbase, 0, 0);    // global_prefetch_b8

    // ---- cur0 = exp(diagonals), masked to k < lens ------------------------
#if HAS_ASYNC_LDS
#pragma unroll
    for (int j = 0; j < 16; ++j) {
        int k = kbase + j;
        if (k < lens) {
            __builtin_amdgcn_global_load_async_to_lds_b32(
                (gptr_t)(dg + k), (lptr_t)&cur[b][k], 0, 0);
        } else {
            cur[b][k] = 0.0f;
        }
    }
    __builtin_amdgcn_s_wait_asynccnt(0);
#pragma unroll
    for (int j = 0; j < 16; ++j) {
        int k = kbase + j;
        if (k < lens) cur[b][k] = expf(cur[b][k]);
    }
#else
#pragma unroll
    for (int j = 0; j < 16; ++j) {
        int k = kbase + j;
        cur[b][k] = (k < lens) ? expf(dg[k]) : 0.0f;
    }
#endif
    __syncthreads();

#pragma unroll
    for (int t = 0; t < DEPTHC; ++t) {
        const int nv = lens - 1 - t;         // masked output width this step

        // ---- phase A: stage cur[k..k+16] into registers (in-place safety)
        float r[17];
#pragma unroll
        for (int j = 0; j <= 16; ++j) {
            int k = kbase + j;
            r[j] = (k < SIZEC) ? cur[b][k] : 0.0f;
        }
        __syncthreads();

        float local = 0.0f;
#pragma unroll
        for (int j = 0; j < 16; ++j) {
            int k = kbase + j;
            if (k < nv) {
                int r0 = k + i + 1 + t;              // never clamps for k < nv
                float num = r[j] * rightp[r0] + r[j + 1] * leftp[k + 1];
                float den = leftp[k] + rightp[r0 + 1];
                float nw  = cst * num / den;
                float lv  = logf(nw);
                cur[b][k] = lv;                      // stash logv in LDS
                local += lv;
            }
        }

        // ---- block reduction of masked logv sum (wave32)
#pragma unroll
        for (int o2 = 16; o2 > 0; o2 >>= 1)
            local += __shfl_xor(local, o2, 32);
        if ((tid & 31) == 0) wsum[tid >> 5] = local;
        __syncthreads();
        if (tid == 0) {
            float s = 0.0f;
#pragma unroll
            for (int w = 0; w < 32; ++w) s += wsum[w];
            mshare = s / ((float)BC * (float)nv);
        }
        __syncthreads();
        const float m = mshare;

        // ---- phase B: normalize
        if (t == DEPTHC - 1) {
            const long oo = 2045L * i - (long)i * (i - 1) / 2;
            float* ob = out + (long)b * OUT_PER_BL + oo;
#pragma unroll
            for (int j = 0; j < 16; ++j) {
                int k = kbase + j;
                if (k < nv) ob[k] = cur[b][k] - m;
            }
        } else {
#pragma unroll
            for (int j = 0; j < 16; ++j) {
                int k = kbase + j;
                if (k < nv) cur[b][k] = expf(cur[b][k] - m);
            }
            __syncthreads();
        }
    }
}

// ---------------------------------------------------------------------------
// Launcher. Inputs (setup_inputs order):
//   0 diagonals (8, TOTAL_IN) f32     3 W_tail (2,12,9) f32   5 const (1,) f32
//   1 tracks    (8, 8, 20480) f32     4 b_tail (2,) f32       6 depth (1,) int
//   2 seq       (8, 4, 20480) f32
// Output: (8, OUT_PER_B) f32.
// Workspace: ypost 327,680 f32 + lr 32,768 f32 = 1.44 MB.
// ---------------------------------------------------------------------------
extern "C" void kernel_launch(void* const* d_in, const int* in_sizes, int n_in,
                              void* d_out, int out_size, void* d_ws, size_t ws_size,
                              hipStream_t stream) {
    (void)in_sizes; (void)n_in; (void)out_size; (void)ws_size;

    const float* diagonals = (const float*)d_in[0];
    const float* tracks    = (const float*)d_in[1];
    const float* seq       = (const float*)d_in[2];
    const float* W_tail    = (const float*)d_in[3];
    const float* b_tail    = (const float*)d_in[4];
    const float* cconst    = (const float*)d_in[5];

    float* out   = (float*)d_out;
    float* ypost = (float*)d_ws;                       // 8*2*20480 floats
    float* lrbuf = ypost + (long)BC * 2 * PPOS;        // 8*2*2048  floats

    // 10240 wave-tiles, 8 waves (= 8 tiles) per 256-thread block
    conv_wmma_kernel<<<(BC * (PPOS / 16)) / 8, 256, 0, stream>>>(
        tracks, seq, W_tail, b_tail, ypost);

    pool_kernel<<<(BC * 2 * SIZEC + 255) / 256, 256, 0, stream>>>(ypost, lrbuf);

    dlem_kernel<<<NDIAG, 1024, 0, stream>>>(diagonals, lrbuf, cconst, out);
}
